// InteractionEncoder_3143916061056
// MI455X (gfx1250) — matrix operand order repaired and checked by reading, not compiled, not run on hardware
//
#include <hip/hip_runtime.h>
#include <hip/hip_bf16.h>
#include <stdint.h>

// ---- problem constants (from reference) ----
#define N_EDGES 500000
#define NT (N_EDGES / 16)   // 31250 tiles of 16 edges
#define L 128
#define LN_EPS 1e-5f

typedef __attribute__((ext_vector_type(16))) __bf16 v16bf;
typedef __attribute__((ext_vector_type(8)))  float  v8f;

union FragU { v16bf v; uint4 q[2]; };

__device__ __forceinline__ unsigned short f2bf(float f) {
  union { float f; uint32_t u; } x; x.f = f;
  uint32_t r = x.u + 0x7FFFu + ((x.u >> 16) & 1u);  // round-to-nearest-even
  return (unsigned short)(r >> 16);
}

// LDS strides (elements), padded for bank-conflict-free fragment access
#define SA    136   // bf16 activation rows (128 cols)
#define SAI   392   // bf16 concat rows (384 cols)
#define SW128 152   // bf16 transposed weight rows, K=128
#define SW384 408   // bf16 transposed weight rows, K=384
#define SP    132   // f32 pre-LN rows (128 cols)

__global__ __launch_bounds__(256)
void fused_interaction_encoder(
    const int*   __restrict__ eidx,
    const float* __restrict__ dx,   const float* __restrict__ eattr,
    const float* __restrict__ va,   const float* __restrict__ vb,   const float* __restrict__ vc,
    const float* __restrict__ svt,  const float* __restrict__ svtm,
    const float* __restrict__ rvt,  const float* __restrict__ rvtm,
    const float* __restrict__ nlat,
    const float* __restrict__ efW1, const float* __restrict__ efB1,
    const float* __restrict__ efW2, const float* __restrict__ efB2,
    const float* __restrict__ efLnB, const float* __restrict__ efLnS,
    const float* __restrict__ eW1,  const float* __restrict__ eB1,
    const float* __restrict__ eW2,  const float* __restrict__ eB2,
    const float* __restrict__ eLnB, const float* __restrict__ eLnS,
    const float* __restrict__ iW1,  const float* __restrict__ iB1,
    const float* __restrict__ iW2,  const float* __restrict__ iB2,
    const float* __restrict__ iLnB, const float* __restrict__ iLnS,
    float* __restrict__ out)
{
  // ---- LDS: weights (bf16, transposed [n][k]) + activation staging ----
  __shared__ unsigned short sWef2[128 * SW128];
  __shared__ unsigned short sWe2 [128 * SW128];
  __shared__ unsigned short sWi2 [128 * SW128];
  __shared__ unsigned short sWi1 [128 * SW384];
  __shared__ unsigned short sAs[16 * SA], sAr[16 * SA], sAe[16 * SA], sAh[16 * SA];
  __shared__ unsigned short sAi[16 * SAI];
  __shared__ float sW1ef[6 * 128], sW1e[2 * 128];
  __shared__ float sB1ef[128], sB2ef[128], sB1e[128], sB2e[128], sB1i[128], sB2i[128];
  __shared__ float sGef[128], sBef[128], sGe[128], sBe[128], sGi[128], sBi[128];
  __shared__ float sSF[16 * 6], sRF[16 * 6], sEF[16 * 2];
  __shared__ float sPreS[16 * SP], sPreR[16 * SP], sPreE[16 * SP];
  __shared__ float sMu[4 * 16], sRs[4 * 16];   // 0:s 1:r 2:e 3:final

  const int tid = threadIdx.x;

  // ---- one-time weight staging (f32 global -> bf16 transposed LDS) ----
  for (int idx = tid; idx < 128 * 128; idx += 256) {
    const int k = idx >> 7, n = idx & 127;
    sWef2[n * SW128 + k] = f2bf(efW2[idx]);
    sWe2 [n * SW128 + k] = f2bf(eW2[idx]);
    sWi2 [n * SW128 + k] = f2bf(iW2[idx]);
  }
  for (int idx = tid; idx < 384 * 128; idx += 256) {
    const int k = idx >> 7, n = idx & 127;
    sWi1[n * SW384 + k] = f2bf(iW1[idx]);
  }
  for (int i = tid; i < 6 * 128; i += 256) sW1ef[i] = efW1[i];
  for (int i = tid; i < 2 * 128; i += 256) sW1e[i]  = eW1[i];
  if (tid < 128) {
    sB1ef[tid] = efB1[tid];  sB2ef[tid] = efB2[tid];
    sB1e[tid]  = eB1[tid];   sB2e[tid]  = eB2[tid];
    sB1i[tid]  = iB1[tid];   sB2i[tid]  = iB2[tid];
    sGef[tid]  = efLnS[tid]; sBef[tid]  = efLnB[tid];
    sGe[tid]   = eLnS[tid];  sBe[tid]   = eLnB[tid];
    sGi[tid]   = iLnS[tid];  sBi[tid]   = iLnB[tid];
  }
  __syncthreads();

  const int wave = tid >> 5, lane = tid & 31;
  const int nn = lane & 15, hh = lane >> 4;
  const int ncol = wave * 16;

  for (int tile = blockIdx.x; tile < NT; tile += gridDim.x) {
    const int e0 = tile * 16;

    // ---- stage A: per-edge geometric features (threads 0..15) ----
    if (tid < 16) {
      const int e = e0 + tid;
      const float a0 = va[3*e], a1 = va[3*e+1], a2 = va[3*e+2];
      const float b0 = vb[3*e], b1 = vb[3*e+1], b2 = vb[3*e+2];
      const float c0 = vc[3*e], c1 = vc[3*e+1], c2 = vc[3*e+2];
      {
        const float v0 = svt[3*e], v1 = svt[3*e+1], v2 = svt[3*e+2];
        sSF[tid*6+0] = a0*v0 + a1*v1 + a2*v2;
        sSF[tid*6+1] = b0*v0 + b1*v1 + b2*v2;
        sSF[tid*6+2] = c0*v0 + c1*v1 + c2*v2;
      }
      {
        const float v0 = svtm[3*e], v1 = svtm[3*e+1], v2 = svtm[3*e+2];
        sSF[tid*6+3] = a0*v0 + a1*v1 + a2*v2;
        sSF[tid*6+4] = b0*v0 + b1*v1 + b2*v2;
        sSF[tid*6+5] = c0*v0 + c1*v1 + c2*v2;
      }
      {
        const float v0 = rvt[3*e], v1 = rvt[3*e+1], v2 = rvt[3*e+2];
        sRF[tid*6+0] = -(a0*v0 + a1*v1 + a2*v2);
        sRF[tid*6+1] = -(b0*v0 + b1*v1 + b2*v2);
        sRF[tid*6+2] = -(c0*v0 + c1*v1 + c2*v2);
      }
      {
        const float v0 = rvtm[3*e], v1 = rvtm[3*e+1], v2 = rvtm[3*e+2];
        sRF[tid*6+3] = -(a0*v0 + a1*v1 + a2*v2);
        sRF[tid*6+4] = -(b0*v0 + b1*v1 + b2*v2);
        sRF[tid*6+5] = -(c0*v0 + c1*v1 + c2*v2);
      }
      {
        const float d0 = dx[3*e], d1 = dx[3*e+1], d2 = dx[3*e+2];
        sEF[tid*2+0] = sqrtf(d0*d0 + d1*d1 + d2*d2);
        sEF[tid*2+1] = eattr[e];
      }
    }
    // ---- stage A2: gather node latents -> sAi cols [128,256) (all threads) ----
    {
      const int r = tid >> 4, cc0 = tid & 15;
      const int sI = eidx[e0 + r];
      const int rI = eidx[N_EDGES + e0 + r];
      const float* ps = nlat + (size_t)sI * L;
      const float* pr = nlat + (size_t)rI * L;
#pragma unroll
      for (int i = 0; i < 8; ++i) {
        const int c = cc0 + 16 * i;
        sAi[r * SAI + 128 + c] = f2bf(ps[c] + pr[c]);
      }
    }
    __syncthreads();

    // ---- stage B: K=6 / K=2 first layers (VALU) -> relu -> bf16 A-matrices ----
    for (int idx = tid; idx < 2048; idx += 256) {
      const int r = idx >> 7, c = idx & 127;
      float hs = sB1ef[c], hr = sB1ef[c];
#pragma unroll
      for (int k = 0; k < 6; ++k) {
        const float w = sW1ef[k * 128 + c];
        hs += sSF[r*6+k] * w;
        hr += sRF[r*6+k] * w;
      }
      sAs[r * SA + c] = f2bf(fmaxf(hs, 0.f));
      sAr[r * SA + c] = f2bf(fmaxf(hr, 0.f));
      const float he = sB1e[c] + sEF[r*2] * sW1e[c] + sEF[r*2+1] * sW1e[128 + c];
      sAe[r * SA + c] = f2bf(fmaxf(he, 0.f));
    }
    __syncthreads();

    // ---- stage C: three 128x128 second layers via WMMA bf16 ----
    {
      v8f accS = {}, accR = {}, accE = {};
#pragma unroll
      for (int kt = 0; kt < 4; ++kt) {
        const int kb = kt * 32;
        FragU a, b;
        const unsigned short* pa = sAs + nn * SA + kb + 8 * hh;
        a.q[0] = *(const uint4*)pa;  a.q[1] = *(const uint4*)(pa + 16);
        const unsigned short* pb = sWef2 + (ncol + nn) * SW128 + kb + 8 * hh;
        b.q[0] = *(const uint4*)pb;  b.q[1] = *(const uint4*)(pb + 16);
        accS = __builtin_amdgcn_wmma_f32_16x16x32_bf16(false, a.v, false, b.v, (short)0, accS, false, false);
        pa = sAr + nn * SA + kb + 8 * hh;
        a.q[0] = *(const uint4*)pa;  a.q[1] = *(const uint4*)(pa + 16);
        accR = __builtin_amdgcn_wmma_f32_16x16x32_bf16(false, a.v, false, b.v, (short)0, accR, false, false);
        pa = sAe + nn * SA + kb + 8 * hh;
        a.q[0] = *(const uint4*)pa;  a.q[1] = *(const uint4*)(pa + 16);
        pb = sWe2 + (ncol + nn) * SW128 + kb + 8 * hh;
        b.q[0] = *(const uint4*)pb;  b.q[1] = *(const uint4*)(pb + 16);
        accE = __builtin_amdgcn_wmma_f32_16x16x32_bf16(false, a.v, false, b.v, (short)0, accE, false, false);
      }
      const float bS = sB2ef[ncol + nn], bE = sB2e[ncol + nn];
#pragma unroll
      for (int j = 0; j < 8; ++j) {
        const int row = j + 8 * hh;
        sPreS[row * SP + ncol + nn] = accS[j] + bS;
        sPreR[row * SP + ncol + nn] = accR[j] + bS;
        sPreE[row * SP + ncol + nn] = accE[j] + bE;
      }
    }
    __syncthreads();

    // ---- stage D: LayerNorm stats for s/r/e (48 threads, one row each) ----
    if (tid < 48) {
      const int which = tid >> 4, r = tid & 15;
      const float* p = (which == 0) ? (sPreS + r * SP)
                     : (which == 1) ? (sPreR + r * SP) : (sPreE + r * SP);
      float s = 0.f, s2 = 0.f;
      for (int c = 0; c < 128; c += 4) {
        const float4 v = *(const float4*)(p + c);
        s  += v.x + v.y + v.z + v.w;
        s2 += v.x*v.x + v.y*v.y + v.z*v.z + v.w*v.w;
      }
      const float mu  = s  * (1.f / 128.f);
      const float var = s2 * (1.f / 128.f) - mu * mu;
      sMu[which * 16 + r] = mu;
      sRs[which * 16 + r] = rsqrtf(var + LN_EPS);
    }
    __syncthreads();

    // ---- stage E: apply LN, build 384-wide concat input (bf16) ----
    for (int idx = tid; idx < 2048; idx += 256) {
      const int r = idx >> 7, c = idx & 127;
      const float sl = (sPreS[r*SP + c] - sMu[r])      * sRs[r]      * sGef[c] + sBef[c];
      const float rl = (sPreR[r*SP + c] - sMu[16 + r]) * sRs[16 + r] * sGef[c] + sBef[c];
      sAi[r * SAI + c] = f2bf(sl + rl);
      const float el = (sPreE[r*SP + c] - sMu[32 + r]) * sRs[32 + r] * sGe[c] + sBe[c];
      sAi[r * SAI + 256 + c] = f2bf(el);
    }
    __syncthreads();

    // ---- stage F: inter layer-1 (K=384) -> relu -> bf16 ----
    {
      v8f acc = {};
#pragma unroll
      for (int kt = 0; kt < 12; ++kt) {
        const int kb = kt * 32;
        FragU a, b;
        const unsigned short* pa = sAi + nn * SAI + kb + 8 * hh;
        a.q[0] = *(const uint4*)pa;  a.q[1] = *(const uint4*)(pa + 16);
        const unsigned short* pb = sWi1 + (ncol + nn) * SW384 + kb + 8 * hh;
        b.q[0] = *(const uint4*)pb;  b.q[1] = *(const uint4*)(pb + 16);
        acc = __builtin_amdgcn_wmma_f32_16x16x32_bf16(false, a.v, false, b.v, (short)0, acc, false, false);
      }
      const float bb = sB1i[ncol + nn];
#pragma unroll
      for (int j = 0; j < 8; ++j) {
        const int row = j + 8 * hh;
        sAh[row * SA + ncol + nn] = f2bf(fmaxf(acc[j] + bb, 0.f));
      }
    }
    __syncthreads();

    // ---- stage G: inter layer-2 (K=128) ----
    {
      v8f acc = {};
#pragma unroll
      for (int kt = 0; kt < 4; ++kt) {
        const int kb = kt * 32;
        FragU a, b;
        const unsigned short* pa = sAh + nn * SA + kb + 8 * hh;
        a.q[0] = *(const uint4*)pa;  a.q[1] = *(const uint4*)(pa + 16);
        const unsigned short* pb = sWi2 + (ncol + nn) * SW128 + kb + 8 * hh;
        b.q[0] = *(const uint4*)pb;  b.q[1] = *(const uint4*)(pb + 16);
        acc = __builtin_amdgcn_wmma_f32_16x16x32_bf16(false, a.v, false, b.v, (short)0, acc, false, false);
      }
      const float bb = sB2i[ncol + nn];
#pragma unroll
      for (int j = 0; j < 8; ++j) {
        const int row = j + 8 * hh;
        sPreS[row * SP + ncol + nn] = acc[j] + bb;   // reuse sPreS
      }
    }
    __syncthreads();

    // ---- stage H: final LayerNorm + coalesced store ----
    if (tid < 16) {
      const float* p = sPreS + tid * SP;
      float s = 0.f, s2 = 0.f;
      for (int c = 0; c < 128; c += 4) {
        const float4 v = *(const float4*)(p + c);
        s  += v.x + v.y + v.z + v.w;
        s2 += v.x*v.x + v.y*v.y + v.z*v.z + v.w*v.w;
      }
      const float mu  = s  * (1.f / 128.f);
      const float var = s2 * (1.f / 128.f) - mu * mu;
      sMu[48 + tid] = mu;
      sRs[48 + tid] = rsqrtf(var + LN_EPS);
    }
    __syncthreads();
    {
      const int c = tid & 127;
      const int r0 = tid >> 7;   // 0 or 1
#pragma unroll
      for (int i = 0; i < 8; ++i) {
        const int r = r0 + 2 * i;
        const float x = (sPreS[r * SP + c] - sMu[48 + r]) * sRs[48 + r] * sGi[c] + sBi[c];
        out[(size_t)(e0 + r) * L + c] = x;
      }
    }
    __syncthreads();
  }
}

extern "C" void kernel_launch(void* const* d_in, const int* in_sizes, int n_in,
                              void* d_out, int out_size, void* d_ws, size_t ws_size,
                              hipStream_t stream) {
  (void)in_sizes; (void)n_in; (void)out_size; (void)d_ws; (void)ws_size;
  // setup_inputs() order; each MLP dict flattens (jax pytree, sorted keys) as:
  //   layers[(W1,b1),(W2,b2)] -> W1,b1,W2,b2 ; then ln_bias ; ln_scale
  const int*   eidx  = (const int*)  d_in[0];
  const float* dx    = (const float*)d_in[1];
  const float* eattr = (const float*)d_in[2];
  const float* va    = (const float*)d_in[3];
  const float* vb    = (const float*)d_in[4];
  const float* vc    = (const float*)d_in[5];
  const float* svt   = (const float*)d_in[6];
  const float* svtm  = (const float*)d_in[7];
  const float* rvt   = (const float*)d_in[8];
  const float* rvtm  = (const float*)d_in[9];
  const float* nlat  = (const float*)d_in[10];
  const float* efW1  = (const float*)d_in[11];
  const float* efB1  = (const float*)d_in[12];
  const float* efW2  = (const float*)d_in[13];
  const float* efB2  = (const float*)d_in[14];
  const float* efLnB = (const float*)d_in[15];
  const float* efLnS = (const float*)d_in[16];
  const float* eW1   = (const float*)d_in[17];
  const float* eB1   = (const float*)d_in[18];
  const float* eW2   = (const float*)d_in[19];
  const float* eB2   = (const float*)d_in[20];
  const float* eLnB  = (const float*)d_in[21];
  const float* eLnS  = (const float*)d_in[22];
  const float* iW1   = (const float*)d_in[23];
  const float* iB1   = (const float*)d_in[24];
  const float* iW2   = (const float*)d_in[25];
  const float* iB2   = (const float*)d_in[26];
  const float* iLnB  = (const float*)d_in[27];
  const float* iLnS  = (const float*)d_in[28];

  dim3 grid(1024), block(256);
  hipLaunchKernelGGL(fused_interaction_encoder, grid, block, 0, stream,
                     eidx, dx, eattr, va, vb, vc, svt, svtm, rvt, rvtm, nlat,
                     efW1, efB1, efW2, efB2, efLnB, efLnS,
                     eW1, eB1, eW2, eB2, eLnB, eLnS,
                     iW1, iB1, iW2, iB2, iLnB, iLnS,
                     (float*)d_out);
}
